// Encoder_6_43413529428769
// MI455X (gfx1250) — compile-verified, matrix operand order checked by reading, not run on hardware
//
#include <hip/hip_runtime.h>
#include <hip/hip_bf16.h>
#include <stdint.h>

typedef __attribute__((ext_vector_type(8)))  __bf16 v8bf;
typedef __attribute__((ext_vector_type(16))) __bf16 v16bf;
typedef __attribute__((ext_vector_type(8)))  float  v8f;
typedef __attribute__((ext_vector_type(4)))  int    v4i;

#define NB   512      // batch
#define LSEQ 192      // sequence length (MAX_LEN_PAD)
#define LPT  196      // padded rows: 2 + 192 + 2
#define CENC 256
#define CP0  288      // 257 in-channels padded to multiple of 32

#if defined(__has_builtin)
#if __has_builtin(__builtin_amdgcn_global_load_async_to_lds_b128) && \
    __has_builtin(__builtin_amdgcn_s_wait_asynccnt)
#define USE_ASYNC 1
#endif
#endif
#ifndef USE_ASYNC
#define USE_ASYNC 0
#endif

// AS(1)/AS(3) typed pointers for the async-DMA builtin (param types per clang
// diagnostic: int4 AS(1)* / LDS side). Formed via integer casts: generic
// addr[63:0] == global addr; generic addr[31:0] == LDS offset (ISA 10.2).
#define GPTR(p) ((__attribute__((address_space(1))) v4i*)(uintptr_t)(p))
#define LPTR(p) ((__attribute__((address_space(3))) v4i*)(uint32_t)(uintptr_t)(p))

// ------------------------------------------------------------------
// Prep: x [B][257][192] f32  ->  act0 [B][196][288] bf16 (zero padded)
// ------------------------------------------------------------------
__global__ void prep_act0(const float* __restrict__ x, __bf16* __restrict__ act)
{
    int b  = blockIdx.x / LPT;
    int tp = blockIdx.x % LPT;
    int c  = threadIdx.x;                  // 0..287
    float v = 0.f;
    if (tp >= 2 && tp < 194 && c < 257)
        v = x[((size_t)b * 257 + c) * LSEQ + (tp - 2)];
    act[((size_t)(b * LPT + tp)) * CP0 + c] = (__bf16)v;
}

// conv weights [256][Cin][5] f32 -> [256][5*CP] bf16, K index = kt*CP + cin
__global__ void pack_convw(const float* __restrict__ w, __bf16* __restrict__ wp,
                           int Cin, int CP)
{
    int idx = blockIdx.x * 256 + threadIdx.x;
    int K   = 5 * CP;
    int m   = idx / K;
    int r   = idx - m * K;
    int kt  = r / CP;
    int cin = r - kt * CP;
    float v = (cin < Cin) ? w[((size_t)m * Cin + cin) * 5 + kt] : 0.f;
    wp[(size_t)m * K + kt * CP + cin] = (__bf16)v;
}

// stack w_ih_f (128x256) and w_ih_b (128x256) -> [256][256] bf16
__global__ void pack_lstmw(const float* __restrict__ wf, const float* __restrict__ wb,
                           __bf16* __restrict__ wp)
{
    int idx = blockIdx.x * 256 + threadIdx.x;
    int m = idx >> 8, k = idx & 255;
    float v = (m < 128) ? wf[(size_t)m * 256 + k] : wb[(size_t)(m - 128) * 256 + k];
    wp[idx] = (__bf16)v;
}

__global__ void prep_lstm_bias(const float* bif, const float* bhf,
                               const float* bib, const float* bhb, float* bl)
{
    int j = threadIdx.x;
    bl[j] = (j < 128) ? (bif[j] + bhf[j]) : (bib[j - 128] + bhb[j - 128]);
}

// ------------------------------------------------------------------
// Implicit-GEMM conv / projection via WMMA bf16 -> f32.
//  act : bf16 [B][LPT][CP]    (rows are padded time, channel contiguous)
//  wp  : bf16 [256][KT*CP]    K index = kt*CP + cin (KT=5 conv, 1 lstm)
//  out : f32  [B][192][256]   = conv(act) + bias
// Block: 256 threads (8 waves). Macro tile M=256 x N=64. K chunks of 32,
// double-buffered through LDS with GLOBAL_LOAD_ASYNC_TO_LDS when available.
// ------------------------------------------------------------------
__global__ __launch_bounds__(256)
void wmma_gemm(const __bf16* __restrict__ act, const __bf16* __restrict__ wp,
               const float* __restrict__ bias, float* __restrict__ out,
               int CP, int KT)
{
    __shared__ __attribute__((aligned(64))) __bf16 lsA[2][256 * 32]; // weights [m][k]
    __shared__ __attribute__((aligned(64))) __bf16 lsB[2][64 * 32];  // activ.  [n][k]

    const int tid    = threadIdx.x;
    const int b      = blockIdx.x / 3;
    const int t0     = (blockIdx.x % 3) * 64;
    const int K      = KT * CP;
    const int Kch    = K >> 5;
    const int wv     = tid >> 5;
    const int lane   = tid & 31;
    const int wave_m = (wv >> 1) * 64;     // 4 waves along M
    const int wave_n = (wv & 1) * 32;      // 2 waves along N
    const int lrow   = lane & 15;
    const int koff   = (lane < 16) ? 0 : 8;
    const int an     = tid >> 2;           // B-staging row (time)
    const int ap     = (tid & 3) * 8;      // B-staging 16B piece

    v8f acc[4][2];
#pragma unroll
    for (int mt = 0; mt < 4; ++mt)
#pragma unroll
        for (int nt = 0; nt < 2; ++nt)
#pragma unroll
            for (int i = 0; i < 8; ++i) acc[mt][nt][i] = 0.f;

    const size_t brow = (size_t)b * LPT;

    int kt_i = 0, cin_i = 0, ka_i = 0;     // next chunk to issue (incremental)
    auto issue = [&](int bufi) {
        const __bf16* asrc = wp + (size_t)tid * K + ka_i;
        __bf16*       adst = &lsA[bufi][tid * 32];
        const __bf16* bsrc = act + (brow + t0 + an + kt_i) * CP + cin_i + ap;
        __bf16*       bdst = &lsB[bufi][an * 32 + ap];
#if USE_ASYNC
        __builtin_amdgcn_global_load_async_to_lds_b128(GPTR(asrc), LPTR(adst), 0, 0);
        __builtin_amdgcn_global_load_async_to_lds_b128(GPTR(asrc), LPTR(adst), 16, 0);
        __builtin_amdgcn_global_load_async_to_lds_b128(GPTR(asrc), LPTR(adst), 32, 0);
        __builtin_amdgcn_global_load_async_to_lds_b128(GPTR(asrc), LPTR(adst), 48, 0);
        __builtin_amdgcn_global_load_async_to_lds_b128(GPTR(bsrc), LPTR(bdst), 0, 0);
#else
        const uint4* s4 = (const uint4*)(const void*)asrc;
        uint4*       d4 = (uint4*)(void*)adst;
        d4[0] = s4[0]; d4[1] = s4[1]; d4[2] = s4[2]; d4[3] = s4[3];
        *(uint4*)(void*)bdst = *(const uint4*)(const void*)bsrc;
#endif
        ka_i  += 32;
        cin_i += 32;
        if (cin_i == CP) { cin_i = 0; ++kt_i; }
    };

    issue(0);                              // prologue: chunk 0 -> buffer 0

    for (int kc = 0; kc < Kch; ++kc) {
        const int cur = kc & 1;
        if (kc + 1 < Kch) {
            issue(cur ^ 1);                // prefetch next chunk into other buffer
#if USE_ASYNC
            __builtin_amdgcn_s_wait_asynccnt(5);   // current chunk landed
#endif
        } else {
#if USE_ASYNC
            __builtin_amdgcn_s_wait_asynccnt(0);
#endif
        }
        __syncthreads();

        v16bf afr[4], bfr[2];
#pragma unroll
        for (int mt = 0; mt < 4; ++mt) {
            const __bf16* rp = &lsA[cur][(wave_m + mt * 16 + lrow) * 32];
            v8bf lo = *(const v8bf*)(const void*)(rp + koff);
            v8bf hi = *(const v8bf*)(const void*)(rp + 16 + koff);
#pragma unroll
            for (int i = 0; i < 8; ++i) { afr[mt][i] = lo[i]; afr[mt][8 + i] = hi[i]; }
        }
#pragma unroll
        for (int nt = 0; nt < 2; ++nt) {
            const __bf16* rp = &lsB[cur][(wave_n + nt * 16 + lrow) * 32];
            v8bf lo = *(const v8bf*)(const void*)(rp + koff);
            v8bf hi = *(const v8bf*)(const void*)(rp + 16 + koff);
#pragma unroll
            for (int i = 0; i < 8; ++i) { bfr[nt][i] = lo[i]; bfr[nt][8 + i] = hi[i]; }
        }
#pragma unroll
        for (int mt = 0; mt < 4; ++mt)
#pragma unroll
            for (int nt = 0; nt < 2; ++nt)
                acc[mt][nt] = __builtin_amdgcn_wmma_f32_16x16x32_bf16(
                    false, afr[mt], false, bfr[nt], (short)0, acc[mt][nt], false, false);

        __syncthreads();
    }

    const int Ml = (lane < 16) ? 0 : 8;
#pragma unroll
    for (int mt = 0; mt < 4; ++mt) {
        const int cbase = wave_m + mt * 16 + Ml;
        float bs[8];
#pragma unroll
        for (int r = 0; r < 8; ++r) bs[r] = bias[cbase + r];
#pragma unroll
        for (int nt = 0; nt < 2; ++nt) {
            const int t = t0 + wave_n + nt * 16 + lrow;
            float* dst = out + ((size_t)(b * LSEQ + t)) * CENC + cbase;
            v8f c = acc[mt][nt];
            float4 w0 = make_float4(c[0] + bs[0], c[1] + bs[1], c[2] + bs[2], c[3] + bs[3]);
            float4 w1 = make_float4(c[4] + bs[4], c[5] + bs[5], c[6] + bs[6], c[7] + bs[7]);
            *(float4*)(void*)dst       = w0;
            *(float4*)(void*)(dst + 4) = w1;
        }
    }
}

// ------------------------------------------------------------------
// GroupNorm(16 groups of 16 ch over T=192) + ReLU, in place on [B][192][256]
// ------------------------------------------------------------------
__global__ __launch_bounds__(256)
void gn_relu(float* __restrict__ h, const float* __restrict__ gw,
             const float* __restrict__ gb)
{
    const int b  = blockIdx.x >> 4;
    const int gp = blockIdx.x & 15;
    const int c0 = gp * 16;
    const int tid = threadIdx.x;

    float s = 0.f, s2 = 0.f;
    for (int e = tid; e < 3072; e += 256) {
        float v = h[((size_t)(b * LSEQ + (e >> 4))) * CENC + c0 + (e & 15)];
        s += v; s2 += v * v;
    }
    __shared__ float rs[256], rq[256];
    rs[tid] = s; rq[tid] = s2;
    __syncthreads();
    for (int off = 128; off; off >>= 1) {
        if (tid < off) { rs[tid] += rs[tid + off]; rq[tid] += rq[tid + off]; }
        __syncthreads();
    }
    const float mean = rs[0] * (1.f / 3072.f);
    const float var  = rq[0] * (1.f / 3072.f) - mean * mean;
    const float rstd = rsqrtf(var + 1e-5f);
    for (int e = tid; e < 3072; e += 256) {
        const int c = c0 + (e & 15);
        const size_t idx = ((size_t)(b * LSEQ + (e >> 4))) * CENC + c;
        float v = (h[idx] - mean) * rstd * gw[c] + gb[c];
        h[idx] = v > 0.f ? v : 0.f;
    }
}

// ------------------------------------------------------------------
// interp_lnr: build per-sample (src index, lambda) map. One thread/sample.
// ------------------------------------------------------------------
__global__ void interp_map(const float* __restrict__ sc, const int* __restrict__ lr,
                           int* __restrict__ midx, float* __restrict__ mlam)
{
    const int b = blockIdx.x * 256 + threadIdx.x;
    if (b >= NB) return;
    int cnt = 0, offset = 0;
    for (int s = 0; s < 7; ++s) {
        const float scale = sc[b * 7 + s] + 0.5f;
        const int   len   = lr[b * 7 + s] + 19;
        const float inv   = 1.f / scale;
        for (int j = 0; j < 64; ++j) {
            const float is  = (float)j * inv;
            const float fl  = floorf(is);
            const float lam = is - fl;
            const float io  = fl + (float)offset;
            if ((fl < (float)(len - 1)) && (io < 191.f)) {
                if (cnt < LSEQ) {
                    int ib = (int)io;
                    ib = ib < 0 ? 0 : (ib > 190 ? 190 : ib);
                    midx[b * LSEQ + cnt] = ib;
                    mlam[b * LSEQ + cnt] = lam;
                }
                ++cnt;
            }
        }
        offset += len;
    }
    for (int p2 = cnt; p2 < LSEQ; ++p2) midx[b * LSEQ + p2] = -1;
}

// Apply warp: gather+lerp from h [B][192][256] f32 -> padded bf16 act [B][196][256]
__global__ __launch_bounds__(256)
void interp_apply(const float* __restrict__ h, const int* __restrict__ midx,
                  const float* __restrict__ mlam, __bf16* __restrict__ actn)
{
    const int b  = blockIdx.x / LPT;
    const int tp = blockIdx.x % LPT;
    const int c  = threadIdx.x;
    float v = 0.f;
    if (tp >= 2 && tp < 194) {
        const int p  = tp - 2;
        const int i0 = midx[b * LSEQ + p];
        if (i0 >= 0) {
            const float lam = mlam[b * LSEQ + p];
            const float* r0 = h + ((size_t)(b * LSEQ + i0)) * CENC + c;
            v = (1.f - lam) * r0[0] + lam * r0[CENC];
        }
    }
    actn[((size_t)(b * LPT + tp)) * CENC + c] = (__bf16)v;
}

// ------------------------------------------------------------------
// LSTM recurrence, both directions fused. Block = 2 waves (f, b), grid = B.
// xw : [B][192][256] f32 (gates i,f,g,o per direction, fwd 0..127 / bwd 128..255)
// out: [B][24][64] f32 (fwd hidden at t=7,15,..  | bwd hidden at t=0,8,..)
// ------------------------------------------------------------------
__global__ __launch_bounds__(64)
void lstm_rec(const float* __restrict__ xw, const float* __restrict__ whf,
              const float* __restrict__ whb, float* __restrict__ out)
{
    __shared__ float sw[2][128][32];
    __shared__ float sh[2][32];
    const int tid = threadIdx.x, b = blockIdx.x;
    const int wv = tid >> 5, j = tid & 31;
    for (int i = tid; i < 128 * 32; i += 64) {
        sw[0][i >> 5][i & 31] = whf[i];
        sw[1][i >> 5][i & 31] = whb[i];
    }
    sh[wv][j] = 0.f;
    float cst = 0.f;
    __syncthreads();
    for (int s = 0; s < LSEQ; ++s) {
        const int t = wv ? (LSEQ - 1 - s) : s;
        const float* g = xw + ((size_t)(b * LSEQ + t)) * CENC + wv * 128;
        float gi = g[j], gf = g[32 + j], gg = g[64 + j], go = g[96 + j];
#pragma unroll 8
        for (int k = 0; k < 32; ++k) {
            const float hk = sh[wv][k];
            gi += hk * sw[wv][j][k];
            gf += hk * sw[wv][32 + j][k];
            gg += hk * sw[wv][64 + j][k];
            go += hk * sw[wv][96 + j][k];
        }
        const float si = 1.f / (1.f + __expf(-gi));
        const float sf = 1.f / (1.f + __expf(-gf));
        const float so = 1.f / (1.f + __expf(-go));
        cst = sf * cst + si * tanhf(gg);
        const float hh = so * tanhf(cst);
        __syncthreads();
        sh[wv][j] = hh;
        __syncthreads();
        if (wv == 0) {
            if ((t & 7) == 7) out[((size_t)(b * 24 + (t >> 3))) * 64 + j] = hh;
        } else {
            if ((t & 7) == 0) out[((size_t)(b * 24 + (t >> 3))) * 64 + 32 + j] = hh;
        }
    }
}

// ------------------------------------------------------------------
extern "C" void kernel_launch(void* const* d_in, const int* in_sizes, int n_in,
                              void* d_out, int out_size, void* d_ws, size_t ws_size,
                              hipStream_t stream)
{
    const float* x      = (const float*)d_in[0];
    const float* scales = (const float*)d_in[1];
    const int*   lenraw = (const int*)  d_in[2];
    const float* cw[3]  = {(const float*)d_in[3],  (const float*)d_in[7],  (const float*)d_in[11]};
    const float* cb[3]  = {(const float*)d_in[4],  (const float*)d_in[8],  (const float*)d_in[12]};
    const float* gg[3]  = {(const float*)d_in[5],  (const float*)d_in[9],  (const float*)d_in[13]};
    const float* gb[3]  = {(const float*)d_in[6],  (const float*)d_in[10], (const float*)d_in[14]};
    const float* w_ih_f = (const float*)d_in[15];
    const float* w_hh_f = (const float*)d_in[16];
    const float* b_ih_f = (const float*)d_in[17];
    const float* b_hh_f = (const float*)d_in[18];
    const float* w_ih_b = (const float*)d_in[19];
    const float* w_hh_b = (const float*)d_in[20];
    const float* b_ih_b = (const float*)d_in[21];
    const float* b_hh_b = (const float*)d_in[22];
    float* out = (float*)d_out;

    char* p = (char*)d_ws;
    auto alloc = [&](size_t bytes) -> char* {
        char* r = p;
        p += (bytes + 255) & ~(size_t)255;
        return r;
    };
    __bf16* actA = (__bf16*)alloc((size_t)NB * LPT * CP0  * 2);
    __bf16* actB = (__bf16*)alloc((size_t)NB * LPT * CENC * 2);
    float*  hbuf = (float*) alloc((size_t)NB * LSEQ * CENC * 4);
    __bf16* wp0  = (__bf16*)alloc((size_t)256 * 1440 * 2);
    __bf16* wp1  = (__bf16*)alloc((size_t)256 * 1280 * 2);
    __bf16* wp2  = (__bf16*)alloc((size_t)256 * 1280 * 2);
    __bf16* wl   = (__bf16*)alloc((size_t)256 * 256 * 2);
    float*  bl   = (float*) alloc(256 * 4);
    int*    midx = (int*)   alloc((size_t)NB * LSEQ * 4);
    float*  mlam = (float*) alloc((size_t)NB * LSEQ * 4);

    // weight / input prep
    prep_act0<<<NB * LPT, 288, 0, stream>>>(x, actA);
    pack_convw<<<1440, 256, 0, stream>>>(cw[0], wp0, 257, CP0);
    pack_convw<<<1280, 256, 0, stream>>>(cw[1], wp1, 256, 256);
    pack_convw<<<1280, 256, 0, stream>>>(cw[2], wp2, 256, 256);
    pack_lstmw<<<256, 256, 0, stream>>>(w_ih_f, w_ih_b, wl);
    prep_lstm_bias<<<1, 256, 0, stream>>>(b_ih_f, b_hh_f, b_ih_b, b_hh_b, bl);

    // layer 0
    wmma_gemm<<<NB * 3, 256, 0, stream>>>(actA, wp0, cb[0], hbuf, CP0, 5);
    gn_relu<<<NB * 16, 256, 0, stream>>>(hbuf, gg[0], gb[0]);
    interp_map<<<2, 256, 0, stream>>>(scales, lenraw, midx, mlam);
    interp_apply<<<NB * LPT, 256, 0, stream>>>(hbuf, midx, mlam, actB);

    // layer 1
    wmma_gemm<<<NB * 3, 256, 0, stream>>>(actB, wp1, cb[1], hbuf, CENC, 5);
    gn_relu<<<NB * 16, 256, 0, stream>>>(hbuf, gg[1], gb[1]);
    interp_map<<<2, 256, 0, stream>>>(scales + NB * 7, lenraw + NB * 7, midx, mlam);
    interp_apply<<<NB * LPT, 256, 0, stream>>>(hbuf, midx, mlam, (__bf16*)actA);

    // layer 2
    wmma_gemm<<<NB * 3, 256, 0, stream>>>((__bf16*)actA, wp2, cb[2], hbuf, CENC, 5);
    gn_relu<<<NB * 16, 256, 0, stream>>>(hbuf, gg[2], gb[2]);
    interp_map<<<2, 256, 0, stream>>>(scales + 2 * NB * 7, lenraw + 2 * NB * 7, midx, mlam);
    interp_apply<<<NB * LPT, 256, 0, stream>>>(hbuf, midx, mlam, actB);

    // LSTM input projection (both directions fused, KT=1) + recurrence
    wmma_gemm<<<NB * 3, 256, 0, stream>>>(actB + 2 * CENC, wl, bl, hbuf, CENC, 1);
    lstm_rec<<<NB, 64, 0, stream>>>(hbuf, w_hh_f, w_hh_b, out);
}
